// GCNE_MLP_combined_28741921145071
// MI455X (gfx1250) — matrix operand order repaired and checked by reading, not compiled, not run on hardware
//
#include <hip/hip_runtime.h>
#include <hip/hip_bf16.h>

typedef __attribute__((ext_vector_type(8)))  _Float16 v8h;
typedef __attribute__((ext_vector_type(16))) _Float16 v16h;
typedef __attribute__((ext_vector_type(8)))  float    v8f;

namespace {
constexpr int kNU = 50000;
constexpr int kNV = 50000;
constexpr int kN  = 100000;       // NU + NV
constexpr int kE  = 1200000;
constexpr int kE2 = kE + kN;      // edges + self loops
constexpr float kNegSlope = 0.2f;
}

// ---------------------------------------------------------------------------
// monotone float <-> uint encoding for atomicMax-based segment max
// ---------------------------------------------------------------------------
__device__ __forceinline__ unsigned fenc(float f) {
  unsigned u = __float_as_uint(f);
  return (u & 0x80000000u) ? ~u : (u | 0x80000000u);
}
__device__ __forceinline__ float fdec(unsigned u) {
  return (u & 0x80000000u) ? __uint_as_float(u & 0x7fffffffu) : __uint_as_float(~u);
}

// ---------------------------------------------------------------------------
// f32 -> f16 conversion (row-major copy)
// ---------------------------------------------------------------------------
__global__ void cvt_kernel(const float* __restrict__ in, _Float16* __restrict__ out, long n) {
  long i = (long)blockIdx.x * blockDim.x + threadIdx.x;
  if (i < n) out[i] = (_Float16)in[i];
}

// f32 [K][N] -> f16 [N][K] (transpose; weights only, done once per launch)
__global__ void cvtT_kernel(const float* __restrict__ in, _Float16* __restrict__ out,
                            int K, int N) {
  long i = (long)blockIdx.x * blockDim.x + threadIdx.x;
  if (i >= (long)K * N) return;
  int k = (int)(i / N), n = (int)(i % N);
  out[(size_t)n * K + k] = (_Float16)in[i];
}

// ---------------------------------------------------------------------------
// WMMA GEMM: C[M,N] = act(A[M,K] * B[K,N] + bias)
//   A  : f16 [M][K] row-major
//   Bt : f16 [N][K] (pre-transposed weights) -> coalesced b128 LDS staging
// Block: 256 threads = 8 waves; tile 64(M) x 64(N); K templated (fully unrolled).
// Wave w: rows 16*(w>>1), cols 32*(w&1): two v8f accumulators, 2 wmma / K-step.
// Double-buffered LDS, one barrier per K-step (syncthreads' dscnt wait covers
// both the previous step's reads and the next buffer's writes).
// A row index is clamped to M-1 so the prefetch is branch-free; padding rows
// are dropped by the guarded epilogue store.
// Fragment layouts per CDNA5 ISA 16-bit A/B VGPR tables.
// ---------------------------------------------------------------------------
template <int K, bool RELU, bool HASBIAS, bool OUTH, bool OUTF>
__global__ __launch_bounds__(256) void gemm_wmma_kernel(
    const _Float16* __restrict__ A, const _Float16* __restrict__ Bt,
    const float* __restrict__ bias, _Float16* __restrict__ Ch,
    float* __restrict__ Cf, int M, int N) {
  constexpr int S = K / 32;
  __shared__ __align__(16) _Float16 As[2][64][40];   // 80B row stride, 16B aligned
  __shared__ __align__(16) _Float16 Bs[2][64][40];   // Bs[n][k]

  const int t    = threadIdx.x;
  const int lane = t & 31;
  const int wave = t >> 5;
  const int bM   = blockIdx.x * 64;
  const int bN   = blockIdx.y * 64;
  const int mrow = (wave >> 1) * 16;   // 0,16,32,48
  const int ncol = (wave & 1) * 32;    // 0,32

  // cooperative staging: each thread one v8h of A and one of Bt per step
  const int aRow  = t >> 2;            // 0..63
  const int aCol  = (t & 3) * 8;       // 0,8,16,24
  const int aRowg = min(bM + aRow, M - 1);   // clamp: branch-free prefetch
  const _Float16* Aptr = A + (size_t)aRowg * K + aCol;
  const _Float16* Bptr = Bt + (size_t)(bN + aRow) * K + aCol;

  {
    v8h av = *(const v8h*)Aptr;
    v8h bv = *(const v8h*)Bptr;
    *(v8h*)&As[0][aRow][aCol] = av;
    *(v8h*)&Bs[0][aRow][aCol] = bv;
  }

  v8f c0 = {}, c1 = {};
  const int m   = mrow + (lane & 15);
  const int klo = (lane >> 4) * 8;     // A: elems 0-7 K=klo.., 8-15 K=16+klo..
  const int n0  = ncol + (lane & 15);
  const int kb  = (lane >> 4) * 8;     // B: elems 0-7 K=kb.., 8-15 K=16+kb..

#pragma unroll
  for (int s = 0; s < S; ++s) {
    __syncthreads();
    const int cur = s & 1;
    // prefetch next K-step into registers (overlaps with wmma below)
    v8h av = {}, bv = {};
    if (s + 1 < S) {
      av = *(const v8h*)(Aptr + (s + 1) * 32);
      bv = *(const v8h*)(Bptr + (s + 1) * 32);
    }
    // A fragment (shared by both wmmas)
    v8h a0 = *(const v8h*)&As[cur][m][klo];
    v8h a1 = *(const v8h*)&As[cur][m][klo + 16];
    v16h a = __builtin_shufflevector(a0, a1, 0, 1, 2, 3, 4, 5, 6, 7,
                                     8, 9, 10, 11, 12, 13, 14, 15);
    // B fragments for the two 16-col tiles
    v8h b0 = *(const v8h*)&Bs[cur][n0][kb];
    v8h b1 = *(const v8h*)&Bs[cur][n0][kb + 16];
    v16h bA = __builtin_shufflevector(b0, b1, 0, 1, 2, 3, 4, 5, 6, 7,
                                      8, 9, 10, 11, 12, 13, 14, 15);
    v8h b2 = *(const v8h*)&Bs[cur][n0 + 16][kb];
    v8h b3 = *(const v8h*)&Bs[cur][n0 + 16][kb + 16];
    v16h bB = __builtin_shufflevector(b2, b3, 0, 1, 2, 3, 4, 5, 6, 7,
                                      8, 9, 10, 11, 12, 13, 14, 15);

    c0 = __builtin_amdgcn_wmma_f32_16x16x32_f16(false, a, false, bA,
                                                (short)0, c0, false, false);
    c1 = __builtin_amdgcn_wmma_f32_16x16x32_f16(false, a, false, bB,
                                                (short)0, c1, false, false);

    if (s + 1 < S) {   // store prefetch into the other buffer
      *(v8h*)&As[1 - cur][aRow][aCol] = av;
      *(v8h*)&Bs[1 - cur][aRow][aCol] = bv;
    }
  }

  // epilogue: VGPR r -> row = r + 8*(lane>=16), col = lane%16
  const int colg = bN + ncol + (lane & 15);
  float bv0 = 0.0f, bv1 = 0.0f;
  if (HASBIAS) { bv0 = bias[colg]; bv1 = bias[colg + 16]; }
#pragma unroll
  for (int r = 0; r < 8; ++r) {
    const int rowg = bM + mrow + r + (lane >> 4) * 8;
    if (rowg < M) {
      float v0 = c0[r] + bv0;
      float v1 = c1[r] + bv1;
      if (RELU) { v0 = v0 > 0.0f ? v0 : 0.0f; v1 = v1 > 0.0f ? v1 : 0.0f; }
      if (OUTF) {
        Cf[(size_t)rowg * N + colg] = v0;
        Cf[(size_t)rowg * N + colg + 16] = v1;
      }
      if (OUTH) {
        Ch[(size_t)rowg * N + colg] = (_Float16)v0;
        Ch[(size_t)rowg * N + colg + 16] = (_Float16)v1;
      }
    }
  }
}

// ---------------------------------------------------------------------------
// edge MLP layer 1: hidden[e][j] = relu(attr[e] * W1[j] + b1[j])  (rank-1)
// ---------------------------------------------------------------------------
__global__ void edge_hidden_kernel(const float* __restrict__ attr,
                                   const float* __restrict__ W1,
                                   const float* __restrict__ b1,
                                   _Float16* __restrict__ out) {
  long i = (long)blockIdx.x * blockDim.x + threadIdx.x;
  if (i >= (long)kE * 64) return;
  int e = (int)(i >> 6), j = (int)(i & 63);
  float v = attr[e] * W1[j] + b1[j];
  out[i] = (_Float16)(v > 0.0f ? v : 0.0f);
}

// ---------------------------------------------------------------------------
// self-loop fill_value='mean': degree + per-node sum of incoming edge features
// ---------------------------------------------------------------------------
__global__ void deg_kernel(const int* __restrict__ ei, float* __restrict__ deg) {
  int e = blockIdx.x * blockDim.x + threadIdx.x;
  if (e < kE) atomicAdd(&deg[ei[kE + e]], 1.0f);
}

__global__ void loopsum_kernel(const int* __restrict__ ei,
                               const _Float16* __restrict__ eh,
                               float* __restrict__ lsum) {
  long i = (long)blockIdx.x * blockDim.x + threadIdx.x;
  if (i >= (long)kE * 64) return;
  int e = (int)(i >> 6), j = (int)(i & 63);
  atomicAdd(&lsum[(size_t)ei[kE + e] * 64 + j], (float)eh[i]);
}

__global__ void loopattr_kernel(const float* __restrict__ lsum,
                                const float* __restrict__ deg,
                                _Float16* __restrict__ lah) {
  long i = (long)blockIdx.x * blockDim.x + threadIdx.x;
  if (i >= (long)kN * 64) return;
  lah[i] = (_Float16)(lsum[i] / fmaxf(deg[i >> 6], 1.0f));
}

// ---------------------------------------------------------------------------
// fold a_e through W_e:  w[c][h] = sum_f We[c][h*F+f] * ae[h][f]   (64 x H)
// ---------------------------------------------------------------------------
__global__ void waeh_kernel(const float* __restrict__ We, const float* __restrict__ ae,
                            float* __restrict__ w, int H, int F) {
  int i = blockIdx.x * blockDim.x + threadIdx.x;
  if (i >= 64 * H) return;
  int c = i / H, h = i % H;
  float s = 0.0f;
  for (int f = 0; f < F; ++f) s += We[(size_t)c * H * F + h * F + f] * ae[h * F + f];
  w[c * H + h] = s;
}

// a_src[n][h], a_dst[n][h] from xw (layout [n][h*F+f])
__global__ void attdot_kernel(const float* __restrict__ xw,
                              const float* __restrict__ a_s, const float* __restrict__ a_d,
                              float* __restrict__ asrc, float* __restrict__ adst,
                              int H, int F) {
  int i = blockIdx.x * blockDim.x + threadIdx.x;
  if (i >= kN * H) return;
  int n = i / H, h = i % H;
  const float* row = xw + (size_t)n * H * F + h * F;
  float ss = 0.0f, sd = 0.0f;
  for (int f = 0; f < F; ++f) { ss += row[f] * a_s[h * F + f]; sd += row[f] * a_d[h * F + f]; }
  asrc[i] = ss; adst[i] = sd;
}

// ae_term[e2][h] = eh_row(e2) . w[:,h]   (eh for real edges, loop_attr for self loops)
__global__ void aeterm_kernel(const _Float16* __restrict__ eh,
                              const _Float16* __restrict__ lah,
                              const float* __restrict__ w,
                              float* __restrict__ aeo, int H) {
  long i = (long)blockIdx.x * blockDim.x + threadIdx.x;
  if (i >= (long)kE2 * H) return;
  int e2 = (int)(i / H), h = (int)(i % H);
  const _Float16* row = (e2 < kE) ? eh + (size_t)e2 * 64 : lah + (size_t)(e2 - kE) * 64;
  float s = 0.0f;
#pragma unroll 8
  for (int c = 0; c < 64; ++c) s += (float)row[c] * w[c * H + h];
  aeo[i] = s;
}

// alpha = leaky_relu(a_src[src]+a_dst[dst]+ae_term) ; segment max via atomicMax
__global__ void alpha_kernel(const int* __restrict__ ei,
                             const float* __restrict__ asrc, const float* __restrict__ adst,
                             const float* __restrict__ aeo,
                             float* __restrict__ alpha, unsigned* __restrict__ mEnc, int H) {
  long i = (long)blockIdx.x * blockDim.x + threadIdx.x;
  if (i >= (long)kE2 * H) return;
  int e2 = (int)(i / H), h = (int)(i % H);
  int s = (e2 < kE) ? ei[e2] : (e2 - kE);
  int d = (e2 < kE) ? ei[kE + e2] : (e2 - kE);
  float a = asrc[(size_t)s * H + h] + adst[(size_t)d * H + h] + aeo[i];
  a = a > 0.0f ? a : kNegSlope * a;
  alpha[i] = a;
  atomicMax(&mEnc[(size_t)d * H + h], fenc(a));
}

// exp(alpha - max) in place + segment sum
__global__ void expsum_kernel(const int* __restrict__ ei,
                              float* __restrict__ alpha,
                              const unsigned* __restrict__ mEnc,
                              float* __restrict__ ssum, int H) {
  long i = (long)blockIdx.x * blockDim.x + threadIdx.x;
  if (i >= (long)kE2 * H) return;
  int e2 = (int)(i / H), h = (int)(i % H);
  int d = (e2 < kE) ? ei[kE + e2] : (e2 - kE);
  float ex = __expf(alpha[i] - fdec(mEnc[(size_t)d * H + h]));
  alpha[i] = ex;
  atomicAdd(&ssum[(size_t)d * H + h], ex);
}

// out[dst] += (ex/sum) * xw[src]   — one block per edge, H*F threads
__global__ void scatter_kernel(const int* __restrict__ ei,
                               const float* __restrict__ ex, const float* __restrict__ ssum,
                               const float* __restrict__ xw, float* __restrict__ out,
                               int H, int F) {
  int e2 = blockIdx.x;
  int t = threadIdx.x;               // h*F + f
  int h = t / F;
  int s = (e2 < kE) ? ei[e2] : (e2 - kE);
  int d = (e2 < kE) ? ei[kE + e2] : (e2 - kE);
  float coef = ex[(size_t)e2 * H + h] / (ssum[(size_t)d * H + h] + 1e-16f);
  atomicAdd(&out[(size_t)d * H * F + t], coef * xw[(size_t)s * H * F + t]);
}

__global__ void fin1_kernel(const float* __restrict__ out1, const float* __restrict__ b,
                            _Float16* __restrict__ x2h) {
  long i = (long)blockIdx.x * blockDim.x + threadIdx.x;
  if (i >= (long)kN * 128) return;
  float v = out1[i] + b[i & 127];
  x2h[i] = (_Float16)(v > 0.0f ? v : 0.0f);
}

__global__ void fin2_kernel(const float* __restrict__ out2, const float* __restrict__ b,
                            float* __restrict__ dout) {
  long i = (long)blockIdx.x * blockDim.x + threadIdx.x;
  if (i >= (long)kN * 64) return;
  dout[i] = out2[i] + b[i & 63];
}

// ---------------------------------------------------------------------------
extern "C" void kernel_launch(void* const* d_in, const int* in_sizes, int n_in,
                              void* d_out, int out_size, void* d_ws, size_t ws_size,
                              hipStream_t stream) {
  (void)in_sizes; (void)n_in; (void)out_size; (void)ws_size;
  const float* xu   = (const float*)d_in[0];
  const float* xv   = (const float*)d_in[1];
  const int*   ei   = (const int*)d_in[2];
  const float* eatt = (const float*)d_in[3];
  const float* uW1 = (const float*)d_in[4];  const float* ub1 = (const float*)d_in[5];
  const float* uW2 = (const float*)d_in[6];  const float* ub2 = (const float*)d_in[7];
  const float* uW3 = (const float*)d_in[8];  const float* ub3 = (const float*)d_in[9];
  const float* vW1 = (const float*)d_in[10]; const float* vb1 = (const float*)d_in[11];
  const float* vW2 = (const float*)d_in[12]; const float* vb2 = (const float*)d_in[13];
  const float* vW3 = (const float*)d_in[14]; const float* vb3 = (const float*)d_in[15];
  const float* eW1 = (const float*)d_in[16]; const float* eb1 = (const float*)d_in[17];
  const float* eW2 = (const float*)d_in[18]; const float* eb2 = (const float*)d_in[19];
  const float* g1W  = (const float*)d_in[20];
  const float* g1as = (const float*)d_in[21]; const float* g1ad = (const float*)d_in[22];
  const float* g1We = (const float*)d_in[23]; const float* g1ae = (const float*)d_in[24];
  const float* g1b  = (const float*)d_in[25];
  const float* g2W  = (const float*)d_in[26];
  const float* g2as = (const float*)d_in[27]; const float* g2ad = (const float*)d_in[28];
  const float* g2We = (const float*)d_in[29]; const float* g2ae = (const float*)d_in[30];
  const float* g2b  = (const float*)d_in[31];

  // ---- workspace carve-out --------------------------------------------------
  char* wsp = (char*)d_ws;
  size_t off = 0;
  auto alloc = [&](size_t bytes) -> char* {
    char* p = wsp + off;
    off = (off + bytes + 255) & ~(size_t)255;
    return p;
  };
  _Float16* xh    = (_Float16*)alloc((size_t)kN * 64 * 2);    // node MLP outputs (f16)
  _Float16* eh    = (_Float16*)alloc((size_t)kE * 64 * 2);    // edge features (f16)
  _Float16* lah   = (_Float16*)alloc((size_t)kN * 64 * 2);    // self-loop edge attr (f16)
  float*    deg   = (float*)alloc((size_t)kN * 4);
  float*    lsum  = (float*)alloc((size_t)kN * 64 * 4);
  float*    xw1   = (float*)alloc((size_t)kN * 128 * 4);
  float*    xw2   = (float*)alloc((size_t)kN * 64 * 4);
  float*    asrc1 = (float*)alloc((size_t)kN * 2 * 4);
  float*    adst1 = (float*)alloc((size_t)kN * 2 * 4);
  float*    asrc2 = (float*)alloc((size_t)kN * 4);
  float*    adst2 = (float*)alloc((size_t)kN * 4);
  float*    wae1  = (float*)alloc(64 * 2 * 4);
  float*    wae2  = (float*)alloc(64 * 4);
  float*    ae1   = (float*)alloc((size_t)kE2 * 2 * 4);
  float*    al1   = (float*)alloc((size_t)kE2 * 2 * 4);
  unsigned* m1    = (unsigned*)alloc((size_t)kN * 2 * 4);
  float*    s1    = (float*)alloc((size_t)kN * 2 * 4);
  float*    out1  = (float*)alloc((size_t)kN * 128 * 4);
  _Float16* x2h   = (_Float16*)alloc((size_t)kN * 128 * 2);
  float*    ae2   = (float*)alloc((size_t)kE2 * 4);
  float*    al2   = (float*)alloc((size_t)kE2 * 4);
  unsigned* m2    = (unsigned*)alloc((size_t)kN * 4);
  float*    s2    = (float*)alloc((size_t)kN * 4);
  float*    out2  = (float*)alloc((size_t)kN * 64 * 4);
  // f16 weight copies, pre-transposed to [N][K]
  _Float16* uW1t = (_Float16*)alloc(256 * 256 * 2);
  _Float16* uW2t = (_Float16*)alloc(256 * 128 * 2);
  _Float16* uW3t = (_Float16*)alloc(128 * 64 * 2);
  _Float16* vW1t = (_Float16*)alloc(256 * 256 * 2);
  _Float16* vW2t = (_Float16*)alloc(256 * 128 * 2);
  _Float16* vW3t = (_Float16*)alloc(128 * 64 * 2);
  _Float16* eW2t = (_Float16*)alloc(64 * 64 * 2);
  _Float16* g1Wt = (_Float16*)alloc(64 * 128 * 2);
  _Float16* g2Wt = (_Float16*)alloc(128 * 64 * 2);
  // transient region (reused): MLP activations, then edge hidden layer
  char* trans = wsp + off;
  _Float16* xin = (_Float16*)trans;                                   // [50000,256] f16
  _Float16* h1  = (_Float16*)(trans + (((size_t)kNU * 256 * 2 + 255) & ~(size_t)255));
  _Float16* h2  = (_Float16*)((char*)h1 + (((size_t)kNU * 256 * 2 + 255) & ~(size_t)255));
  _Float16* ehid = (_Float16*)trans;                                  // [E,64] f16 (aliases MLP bufs)

  auto cvt = [&](const float* src, _Float16* dst, long n) {
    cvt_kernel<<<dim3((unsigned)((n + 255) / 256)), dim3(256), 0, stream>>>(src, dst, n);
  };
  auto cvtT = [&](const float* src, _Float16* dst, int K, int N) {
    long n = (long)K * N;
    cvtT_kernel<<<dim3((unsigned)((n + 255) / 256)), dim3(256), 0, stream>>>(src, dst, K, N);
  };
  auto g1d = [](long n) { return dim3((unsigned)((n + 255) / 256)); };
  auto grid = [](int M, int N) { return dim3((unsigned)((M + 63) / 64), (unsigned)(N / 64)); };

  // ---- zero accumulators ----------------------------------------------------
  hipMemsetAsync(deg, 0, (size_t)kN * 4, stream);
  hipMemsetAsync(lsum, 0, (size_t)kN * 64 * 4, stream);
  hipMemsetAsync(m1, 0, (size_t)kN * 2 * 4, stream);
  hipMemsetAsync(s1, 0, (size_t)kN * 2 * 4, stream);
  hipMemsetAsync(out1, 0, (size_t)kN * 128 * 4, stream);
  hipMemsetAsync(m2, 0, (size_t)kN * 4, stream);
  hipMemsetAsync(s2, 0, (size_t)kN * 4, stream);
  hipMemsetAsync(out2, 0, (size_t)kN * 64 * 4, stream);

  // ---- weights to f16 (transposed) ------------------------------------------
  cvtT(uW1, uW1t, 256, 256); cvtT(uW2, uW2t, 256, 128); cvtT(uW3, uW3t, 128, 64);
  cvtT(vW1, vW1t, 256, 256); cvtT(vW2, vW2t, 256, 128); cvtT(vW3, vW3t, 128, 64);
  cvtT(eW2, eW2t, 64, 64);   cvtT(g1W, g1Wt, 64, 128);  cvtT(g2W, g2Wt, 128, 64);

  // ---- node MLPs (WMMA GEMM chain) -----------------------------------------
  cvt(xu, xin, (long)kNU * 256);
  gemm_wmma_kernel<256, true, true, true, false><<<grid(kNU, 256), 256, 0, stream>>>(
      xin, uW1t, ub1, h1, nullptr, kNU, 256);
  gemm_wmma_kernel<256, true, true, true, false><<<grid(kNU, 128), 256, 0, stream>>>(
      h1, uW2t, ub2, h2, nullptr, kNU, 128);
  gemm_wmma_kernel<128, false, true, true, false><<<grid(kNU, 64), 256, 0, stream>>>(
      h2, uW3t, ub3, xh, nullptr, kNU, 64);
  cvt(xv, xin, (long)kNV * 256);
  gemm_wmma_kernel<256, true, true, true, false><<<grid(kNV, 256), 256, 0, stream>>>(
      xin, vW1t, vb1, h1, nullptr, kNV, 256);
  gemm_wmma_kernel<256, true, true, true, false><<<grid(kNV, 128), 256, 0, stream>>>(
      h1, vW2t, vb2, h2, nullptr, kNV, 128);
  gemm_wmma_kernel<128, false, true, true, false><<<grid(kNV, 64), 256, 0, stream>>>(
      h2, vW3t, vb3, xh + (size_t)kNU * 64, nullptr, kNV, 64);

  // ---- edge MLP -------------------------------------------------------------
  edge_hidden_kernel<<<g1d((long)kE * 64), 256, 0, stream>>>(eatt, eW1, eb1, ehid);
  gemm_wmma_kernel<64, false, true, true, false><<<grid(kE, 64), 256, 0, stream>>>(
      ehid, eW2t, eb2, eh, nullptr, kE, 64);

  // ---- self-loop mean edge attr --------------------------------------------
  deg_kernel<<<g1d(kE), 256, 0, stream>>>(ei, deg);
  loopsum_kernel<<<g1d((long)kE * 64), 256, 0, stream>>>(ei, eh, lsum);
  loopattr_kernel<<<g1d((long)kN * 64), 256, 0, stream>>>(lsum, deg, lah);

  // ---- GAT conv 1 (heads=2, F=64) ------------------------------------------
  waeh_kernel<<<1, 128, 0, stream>>>(g1We, g1ae, wae1, 2, 64);
  gemm_wmma_kernel<64, false, false, false, true><<<grid(kN, 128), 256, 0, stream>>>(
      xh, g1Wt, nullptr, nullptr, xw1, kN, 128);
  attdot_kernel<<<g1d((long)kN * 2), 256, 0, stream>>>(xw1, g1as, g1ad, asrc1, adst1, 2, 64);
  aeterm_kernel<<<g1d((long)kE2 * 2), 256, 0, stream>>>(eh, lah, wae1, ae1, 2);
  alpha_kernel<<<g1d((long)kE2 * 2), 256, 0, stream>>>(ei, asrc1, adst1, ae1, al1, m1, 2);
  expsum_kernel<<<g1d((long)kE2 * 2), 256, 0, stream>>>(ei, al1, m1, s1, 2);
  scatter_kernel<<<dim3(kE2), dim3(128), 0, stream>>>(ei, al1, s1, xw1, out1, 2, 64);
  fin1_kernel<<<g1d((long)kN * 128), 256, 0, stream>>>(out1, g1b, x2h);

  // ---- GAT conv 2 (heads=1, F=64) ------------------------------------------
  waeh_kernel<<<1, 64, 0, stream>>>(g2We, g2ae, wae2, 1, 64);
  gemm_wmma_kernel<128, false, false, false, true><<<grid(kN, 64), 256, 0, stream>>>(
      x2h, g2Wt, nullptr, nullptr, xw2, kN, 64);
  attdot_kernel<<<g1d((long)kN), 256, 0, stream>>>(xw2, g2as, g2ad, asrc2, adst2, 1, 64);
  aeterm_kernel<<<g1d((long)kE2), 256, 0, stream>>>(eh, lah, wae2, ae2, 1);
  alpha_kernel<<<g1d((long)kE2), 256, 0, stream>>>(ei, asrc2, adst2, ae2, al2, m2, 1);
  expsum_kernel<<<g1d((long)kE2), 256, 0, stream>>>(ei, al2, m2, s2, 1);
  scatter_kernel<<<dim3(kE2), dim3(64), 0, stream>>>(ei, al2, s2, xw2, out2, 1, 64);
  fin2_kernel<<<g1d((long)kN * 64), 256, 0, stream>>>(out2, g2b, (float*)d_out);
}
